// MultiheadAttention_63891933495488
// MI455X (gfx1250) — compile-verified, matrix operand order checked by reading, not run on hardware
//
#include <hip/hip_runtime.h>
#include <hip/hip_bf16.h>
#include <math.h>

// ---------------------------------------------------------------------------
// MI455X (gfx1250) fused causal MHA with relative-position bias.
// All GEMM work on v_wmma_f32_16x16x32_bf16 (bf16 in, f32 accum).
// Round 2: vectorized staging (global_load_b128 + packed v8bf LDS stores),
// 128x64 GEMM tiles (4 WMMA per wave per K-step sharing one B fragment).
// ---------------------------------------------------------------------------

typedef __attribute__((ext_vector_type(16))) __bf16 v16bf;
typedef __attribute__((ext_vector_type(8)))  __bf16 v8bf;
typedef __attribute__((ext_vector_type(8)))  float  v8f;

#define NUM_HEADS 16
#define EMBED     1024
#define HEAD      64
#define SEQ       2048
#define BATCH     2
#define MROWS     (BATCH * SEQ)   // 4096
#define MAX_DIST  4

// Load a 16x32 bf16 WMMA fragment (A layout; B of a row-major-over-K matrix
// uses the identical pattern with row := output column).
// ISA layout: lane L holds row = L&15; VGPR0..3 hold K = kb..kb+7,
// VGPR4..7 hold K = 16+kb..16+kb+7, where kb = (L>>4)*8.
static __device__ __forceinline__ v16bf load_frag(const __bf16* tile, int stride, int lane) {
  const int row = lane & 15;
  const int kb  = (lane >> 4) << 3;
  const __bf16* p = tile + row * stride + kb;
  v16bf f;
#pragma unroll
  for (int i = 0; i < 8; ++i) f[i] = p[i];
#pragma unroll
  for (int i = 0; i < 8; ++i) f[8 + i] = p[16 + i];
  return f;
}

static __device__ __forceinline__ v8f wmma_bf16(v16bf a, v16bf b, v8f c) {
  // (neg_a, A, neg_b, B, c_mod, C, reuse_a, reuse_b)
  return __builtin_amdgcn_wmma_f32_16x16x32_bf16(false, a, false, b, (short)0, c, false, false);
}

// Read 8 consecutive f32 (two global_load_b128), convert, one 16B LDS store.
static __device__ __forceinline__ void stage8_f32(__bf16* dst, const float* src) {
  const float4 x0 = *(const float4*)(src);
  const float4 x1 = *(const float4*)(src + 4);
  v8bf t;
  t[0] = (__bf16)x0.x; t[1] = (__bf16)x0.y; t[2] = (__bf16)x0.z; t[3] = (__bf16)x0.w;
  t[4] = (__bf16)x1.x; t[5] = (__bf16)x1.y; t[6] = (__bf16)x1.z; t[7] = (__bf16)x1.w;
  *(v8bf*)dst = t;
}

// Store one 16x16 f32 C fragment into the bf16 head-major tensor [b][h][s][d].
static __device__ __forceinline__ void store_head_tile(v8f acc, int m0, int n0, int lane,
                                                       float scale, __bf16* __restrict__ Y) {
  const int n = n0 + (lane & 15);
  const int h = n >> 6;           // HEAD = 64
  const int d = n & 63;
#pragma unroll
  for (int j = 0; j < 8; ++j) {
    const int m  = m0 + j + ((lane >> 4) << 3);
    const int bb = m >> 11;       // SEQ = 2048
    const int sp = m & 2047;
    Y[(((size_t)bb * NUM_HEADS + h) * SEQ + sp) * HEAD + d] = (__bf16)(acc[j] * scale);
  }
}

// ---------------------------------------------------------------------------
// Kernel 1: Y[b][h][s][d] = scale * (X @ W^T), X f32 (4096 x 1024), W f32 (1024 x 1024)
// 128x64 tile / 256-thread block (8 waves, 4 C-subtiles per wave), K-step 32.
// ---------------------------------------------------------------------------
__global__ __launch_bounds__(256) void proj_head_gemm(const float* __restrict__ X,
                                                      const float* __restrict__ W,
                                                      __bf16* __restrict__ Y,
                                                      float scale) {
  __shared__ __bf16 As[128][40];  // +8 pad: conflict-free, 16B-aligned frags
  __shared__ __bf16 Bs[64][40];
  const int tid   = threadIdx.x;
  const int lane  = tid & 31;
  const int wave  = tid >> 5;
  const int nsub  = wave & 3;
  const int mbase = (wave >> 2) << 2;     // 0 or 4 (4 m-subtiles per wave)
  const int m0 = blockIdx.y * 128;
  const int n0 = blockIdx.x * 64;

  v8f acc[4] = {};
  for (int kk = 0; kk < EMBED; kk += 32) {
    // A tile 128x32: 512 8-elem chunks, 2 per thread
#pragma unroll
    for (int i = 0; i < 2; ++i) {
      const int chunk = tid + (i << 8);
      const int r = chunk >> 2, c = (chunk & 3) << 3;
      stage8_f32(&As[r][c], &X[(size_t)(m0 + r) * EMBED + kk + c]);
    }
    // B tile 64x32: 256 chunks, 1 per thread
    {
      const int r = tid >> 2, c = (tid & 3) << 3;
      stage8_f32(&Bs[r][c], &W[(size_t)(n0 + r) * EMBED + kk + c]);
    }
    if (kk + 32 < EMBED)                  // global_prefetch_b8 of next K slab
      __builtin_prefetch(&X[(size_t)(m0 + (tid >> 1)) * EMBED + kk + 32], 0, 1);
    __syncthreads();

    const v16bf bf = load_frag(&Bs[nsub * 16][0], 40, lane);
#pragma unroll
    for (int s = 0; s < 4; ++s) {
      const v16bf a = load_frag(&As[(mbase + s) * 16][0], 40, lane);
      acc[s] = wmma_bf16(a, bf, acc[s]);
    }
    __syncthreads();
  }
#pragma unroll
  for (int s = 0; s < 4; ++s)
    store_head_tile(acc[s], m0 + (mbase + s) * 16, n0 + nsub * 16, lane, scale, Y);
}

// ---------------------------------------------------------------------------
// Kernel 2: flash attention per (b,h,q-tile of 64) with causal mask and
// clipped relative-position bias. Q already carries 1/sqrt(HEAD).
// ---------------------------------------------------------------------------
__global__ __launch_bounds__(256) void attn_kernel(const __bf16* __restrict__ Qh,
                                                   const __bf16* __restrict__ Kh,
                                                   const __bf16* __restrict__ Vh,
                                                   const float*  __restrict__ Pp,   // (HEAD, 9)
                                                   __bf16* __restrict__ O) {        // (b, s, h*64+d)
  __shared__ __bf16 Qs[64][72];
  __shared__ __bf16 Ks[64][72];
  __shared__ __bf16 Vts[64][72];            // transposed: Vts[d][k]
  __shared__ __bf16 Ps[64][72];
  __shared__ float  Ss[64][66];
  __shared__ float  bias_s[64][9];
  __shared__ float  pp_s[HEAD * 9];
  __shared__ float  m_s[64], l_s[64], sc_s[64];

  const int tid   = threadIdx.x;
  const int lane  = tid & 31;
  const int wave  = tid >> 5;
  const int nsub  = wave & 3;
  const int msub0 = (wave >> 2) << 1;

  const int qt = blockIdx.x;
  const int bh = blockIdx.y;                // b*16 + h
  const int b  = bh >> 4;
  const int h  = bh & 15;
  const int q0 = qt * 64;

  const __bf16* qp = Qh + (size_t)bh * SEQ * HEAD;
  const __bf16* kp = Kh + (size_t)bh * SEQ * HEAD;
  const __bf16* vp = Vh + (size_t)bh * SEQ * HEAD;

  // stage Q tile (64x64 bf16): 512 8-elem chunks, 2 per thread, b128 traffic
#pragma unroll
  for (int i = 0; i < 2; ++i) {
    const int chunk = tid + (i << 8);
    const int r = chunk >> 3, c = (chunk & 7) << 3;
    *(v8bf*)&Qs[r][c] = *(const v8bf*)&qp[(size_t)(q0 + r) * HEAD + c];
  }
  for (int i = tid; i < HEAD * 9; i += 256) pp_s[i] = Pp[i];
  __syncthreads();

  if (tid < 64) {                           // per-row bias buckets: qh @ p_params
    m_s[tid] = -3.0e38f;
    l_s[tid] = 0.0f;
#pragma unroll
    for (int r = 0; r < 9; ++r) {
      float acc = 0.0f;
      for (int d = 0; d < HEAD; ++d) acc += (float)Qs[tid][d] * pp_s[d * 9 + r];
      bias_s[tid][r] = acc;
    }
  }
  __syncthreads();

  v8f o0 = {}, o1 = {};
  for (int kt = 0; kt <= qt; ++kt) {        // causal: only lower-triangular tiles
    // stage K (row-major) and V (transposed into Vts[d][k])
#pragma unroll
    for (int i = 0; i < 2; ++i) {
      const int chunk = tid + (i << 8);
      const int r = chunk >> 3, c = (chunk & 7) << 3;
      *(v8bf*)&Ks[r][c] = *(const v8bf*)&kp[(size_t)(kt * 64 + r) * HEAD + c];
      const v8bf vv = *(const v8bf*)&vp[(size_t)(kt * 64 + r) * HEAD + c];
#pragma unroll
      for (int j = 0; j < 8; ++j) Vts[c + j][r] = vv[j];
    }
    __syncthreads();

    {                                       // S = Q @ K^T  (contraction d = 64)
      const v16bf bk0  = load_frag(&Ks[nsub * 16][0],  72, lane);
      const v16bf bk1  = load_frag(&Ks[nsub * 16][32], 72, lane);
      const v16bf aq00 = load_frag(&Qs[msub0 * 16][0],  72, lane);
      const v16bf aq01 = load_frag(&Qs[msub0 * 16][32], 72, lane);
      const v16bf aq10 = load_frag(&Qs[(msub0 + 1) * 16][0],  72, lane);
      const v16bf aq11 = load_frag(&Qs[(msub0 + 1) * 16][32], 72, lane);
      v8f s0 = {}, s1 = {};
      s0 = wmma_bf16(aq00, bk0, s0); s0 = wmma_bf16(aq01, bk1, s0);
      s1 = wmma_bf16(aq10, bk0, s1); s1 = wmma_bf16(aq11, bk1, s1);
      const int n = nsub * 16 + (lane & 15);
#pragma unroll
      for (int j = 0; j < 8; ++j) {
        Ss[msub0 * 16 + j + ((lane >> 4) << 3)][n]       = s0[j];
        Ss[(msub0 + 1) * 16 + j + ((lane >> 4) << 3)][n] = s1[j];
      }
    }
    __syncthreads();

    if (tid < 64) {                         // mask + bias + online softmax row pass
      const int r    = tid;
      const int qpos = q0 + r;
      const float mold = m_s[r];
      float tmax = mold;
      for (int c = 0; c < 64; ++c) {
        const int kpos = kt * 64 + c;
        float s = Ss[r][c];
        if (kpos > qpos) {
          s = -1.0e30f;                     // NEG_INF per reference
        } else {
          int dist = qpos - kpos;
          if (dist > MAX_DIST) dist = MAX_DIST;
          s += bias_s[r][dist + MAX_DIST];
        }
        Ss[r][c] = s;
        tmax = fmaxf(tmax, s);
      }
      const float mnew = tmax;
      const float sc   = __expf(mold - mnew);
      float lsum = 0.0f;
      for (int c = 0; c < 64; ++c) {
        const float p = __expf(Ss[r][c] - mnew);
        lsum += p;
        Ps[r][c] = (__bf16)p;
      }
      m_s[r]  = mnew;
      l_s[r]  = l_s[r] * sc + lsum;
      sc_s[r] = sc;
    }
    __syncthreads();

#pragma unroll
    for (int j = 0; j < 8; ++j) {           // rescale running O by exp(mold-mnew)
      o0[j] *= sc_s[msub0 * 16 + j + ((lane >> 4) << 3)];
      o1[j] *= sc_s[(msub0 + 1) * 16 + j + ((lane >> 4) << 3)];
    }
    {                                       // O += P @ V  (contraction k = 64)
      const v16bf bv0  = load_frag(&Vts[nsub * 16][0],  72, lane);
      const v16bf bv1  = load_frag(&Vts[nsub * 16][32], 72, lane);
      const v16bf pa00 = load_frag(&Ps[msub0 * 16][0],  72, lane);
      const v16bf pa01 = load_frag(&Ps[msub0 * 16][32], 72, lane);
      const v16bf pa10 = load_frag(&Ps[(msub0 + 1) * 16][0],  72, lane);
      const v16bf pa11 = load_frag(&Ps[(msub0 + 1) * 16][32], 72, lane);
      o0 = wmma_bf16(pa00, bv0, o0); o0 = wmma_bf16(pa01, bv1, o0);
      o1 = wmma_bf16(pa10, bv0, o1); o1 = wmma_bf16(pa11, bv1, o1);
    }
    __syncthreads();                        // protect LDS reuse next k-tile
  }

  const int n = nsub * 16 + (lane & 15);    // d within head
#pragma unroll
  for (int j = 0; j < 8; ++j) {             // normalize by l, store (b, q, h*64+d)
    const int r0 = msub0 * 16 + j + ((lane >> 4) << 3);
    const int r1 = (msub0 + 1) * 16 + j + ((lane >> 4) << 3);
    O[(size_t)(b * SEQ + q0 + r0) * EMBED + h * HEAD + n] = (__bf16)(o0[j] / l_s[r0]);
    O[(size_t)(b * SEQ + q0 + r1) * EMBED + h * HEAD + n] = (__bf16)(o1[j] / l_s[r1]);
  }
}

// ---------------------------------------------------------------------------
// Kernel 3: out(f32, 4096 x 1024) = A(bf16) @ Wo^T (f32 -> bf16 convert)
// 128x64 tile / block, same scheme as kernel 1.
// ---------------------------------------------------------------------------
__global__ __launch_bounds__(256) void out_proj_gemm(const __bf16* __restrict__ X,
                                                     const float* __restrict__ W,
                                                     float* __restrict__ out) {
  __shared__ __bf16 As[128][40];
  __shared__ __bf16 Bs[64][40];
  const int tid   = threadIdx.x;
  const int lane  = tid & 31;
  const int wave  = tid >> 5;
  const int nsub  = wave & 3;
  const int mbase = (wave >> 2) << 2;
  const int m0 = blockIdx.y * 128;
  const int n0 = blockIdx.x * 64;

  v8f acc[4] = {};
  for (int kk = 0; kk < EMBED; kk += 32) {
#pragma unroll
    for (int i = 0; i < 2; ++i) {           // A tile 128x32, bf16 direct copy
      const int chunk = tid + (i << 8);
      const int r = chunk >> 2, c = (chunk & 3) << 3;
      *(v8bf*)&As[r][c] = *(const v8bf*)&X[(size_t)(m0 + r) * EMBED + kk + c];
    }
    {                                       // B tile 64x32, f32 -> bf16
      const int r = tid >> 2, c = (tid & 3) << 3;
      stage8_f32(&Bs[r][c], &W[(size_t)(n0 + r) * EMBED + kk + c]);
    }
    __syncthreads();
    const v16bf bf = load_frag(&Bs[nsub * 16][0], 40, lane);
#pragma unroll
    for (int s = 0; s < 4; ++s) {
      const v16bf a = load_frag(&As[(mbase + s) * 16][0], 40, lane);
      acc[s] = wmma_bf16(a, bf, acc[s]);
    }
    __syncthreads();
  }
  const int n = n0 + nsub * 16 + (lane & 15);
#pragma unroll
  for (int s = 0; s < 4; ++s) {
#pragma unroll
    for (int j = 0; j < 8; ++j) {
      const int m = m0 + (mbase + s) * 16 + j + ((lane >> 4) << 3);
      out[(size_t)m * EMBED + n] = acc[s][j];
    }
  }
}

// ---------------------------------------------------------------------------
extern "C" void kernel_launch(void* const* d_in, const int* in_sizes, int n_in,
                              void* d_out, int out_size, void* d_ws, size_t ws_size,
                              hipStream_t stream) {
  (void)in_sizes; (void)n_in; (void)out_size; (void)ws_size;
  const float* k  = (const float*)d_in[0];
  const float* v  = (const float*)d_in[1];
  const float* q  = (const float*)d_in[2];
  const float* Wk = (const float*)d_in[3];
  const float* Wv = (const float*)d_in[4];
  const float* Wq = (const float*)d_in[5];
  const float* Wo = (const float*)d_in[6];
  const float* pp = (const float*)d_in[7];
  float* out = (float*)d_out;

  // Workspace: 4 bf16 tensors of B*H*S*D = 4Mi elements (8 MB each, 32 MB total)
  const size_t headElems = (size_t)BATCH * NUM_HEADS * SEQ * HEAD;
  __bf16* qh = (__bf16*)d_ws;
  __bf16* kh = qh + headElems;
  __bf16* vh = kh + headElems;
  __bf16* ah = vh + headElems;

  const dim3 blk(256);
  const dim3 gemmGrid(EMBED / 64, MROWS / 128);      // (16, 32)
  const float qscale = 0.125f;                       // 1/sqrt(HEAD); covers scores AND bias

  proj_head_gemm<<<gemmGrid, blk, 0, stream>>>(q, Wq, qh, qscale);
  proj_head_gemm<<<gemmGrid, blk, 0, stream>>>(k, Wk, kh, 1.0f);
  proj_head_gemm<<<gemmGrid, blk, 0, stream>>>(v, Wv, vh, 1.0f);
  attn_kernel<<<dim3(SEQ / 64, BATCH * NUM_HEADS), blk, 0, stream>>>(qh, kh, vh, pp, ah);
  out_proj_gemm<<<gemmGrid, blk, 0, stream>>>(ah, Wo, out);
}
